// MemoryEfficientAttention_21663815041751
// MI455X (gfx1250) — matrix operand order repaired and checked by reading, not compiled, not run on hardware
//
#include <hip/hip_runtime.h>
#include <cstdint>

// ---------------- types ----------------
typedef __bf16 bf16;
typedef __attribute__((ext_vector_type(16))) __bf16 v16bf;
typedef __attribute__((ext_vector_type(8)))  __bf16 v8bf;
typedef __attribute__((ext_vector_type(8)))  float  v8f;

#define B_    2
#define S_    2048
#define DIM_  1024
#define H_    16
#define HD_   64
#define SCALE_ 0.125f   // HD^-0.5

union FragBF { v16bf f; v8bf h[2]; };

// D = A(16x32 bf16) * B(32x16 bf16) + C(16x16 f32)
__device__ inline v8f wmma_bf16(v16bf a, v16bf b, v8f c) {
  return __builtin_amdgcn_wmma_f32_16x16x32_bf16(
      /*neg_a=*/false, a, /*neg_b=*/false, b,
      /*c_mod=*/(short)0, c, /*reuse_a=*/false, /*reuse_b=*/false);
}

// A-fragment 16x32 (row-major A, leading dim ldk, base at A[row0][k0]).
// lane m=L&15, half=L>>4; VGPR0-3 = K[half*8..+7], VGPR4-7 = K[16+half*8..+7]
__device__ inline v16bf load_afrag(const bf16* base, int ldk, int lane) {
  int m = lane & 15, half = lane >> 4;
  const bf16* p = base + (size_t)m * ldk + half * 8;
  FragBF fr;
  fr.h[0] = *(const v8bf*)(p);
  fr.h[1] = *(const v8bf*)(p + 16);
  return fr.f;
}

// B-fragment 32x16, sourced from Bt[n][k] (column-major B), leading dim ldk.
// lane n=L&15, half=L>>4; VGPR0-3 = K[half*8..+7], VGPR4-7 = K[16+half*8..+7] of column n
__device__ inline v16bf load_bfrag(const bf16* base, int ldk, int lane) {
  int n = lane & 15, half = lane >> 4;
  const bf16* p = base + (size_t)n * ldk + half * 8;
  FragBF fr;
  fr.h[0] = *(const v8bf*)(p);
  fr.h[1] = *(const v8bf*)(p + 16);
  return fr.f;
}

// ---------------- conversion kernels ----------------
__global__ void cvt_f32_bf16(const float* __restrict__ x, bf16* __restrict__ y, int n) {
  int i = blockIdx.x * 256 + threadIdx.x;
  if (i < n) y[i] = (bf16)x[i];
}

// yt[n][k] = w[k][n]   (w is rows x cols row-major; yt is cols x rows)
__global__ void cvt_transpose_bf16(const float* __restrict__ w, bf16* __restrict__ yt,
                                   int rows, int cols) {
  int i = blockIdx.x * 256 + threadIdx.x;
  if (i < rows * cols) {
    int nIdx = i / rows;
    int k    = i % rows;
    yt[i] = (bf16)w[(size_t)k * cols + nIdx];
  }
}

// ---------------- shared GEMM mainloop: 32x64 tile per wave, double-buffered ----
// acc[t*2+ar] accumulates rows [r0+ar*16, +16) x cols [c0+t*16, +16)
__device__ inline void gemm_mainloop_32x64(const bf16* __restrict__ Abase,
                                           const bf16* __restrict__ Btbase,
                                           int lane, v8f acc[8]) {
  const int K = DIM_;
  v16bf a0 = load_afrag(Abase, K, lane);
  v16bf a1 = load_afrag(Abase + 16 * K, K, lane);
  v16bf b0 = load_bfrag(Btbase +  0 * K, K, lane);
  v16bf b1 = load_bfrag(Btbase + 16 * K, K, lane);
  v16bf b2 = load_bfrag(Btbase + 32 * K, K, lane);
  v16bf b3 = load_bfrag(Btbase + 48 * K, K, lane);

  for (int k0 = 0; k0 < K; k0 += 32) {
    int kn = (k0 + 32) & (K - 1);     // wrapped: branch-free double buffer
    int kp = (k0 + 128) & (K - 1);    // prefetch distance: 4 k-steps
    __builtin_prefetch(Btbase + (size_t)(lane & 15) * K + kp, 0, 0);
    // issue next-iteration loads before consuming current fragments
    v16bf na0 = load_afrag(Abase + kn, K, lane);
    v16bf na1 = load_afrag(Abase + 16 * K + kn, K, lane);
    v16bf nb0 = load_bfrag(Btbase +  0 * K + kn, K, lane);
    v16bf nb1 = load_bfrag(Btbase + 16 * K + kn, K, lane);
    v16bf nb2 = load_bfrag(Btbase + 32 * K + kn, K, lane);
    v16bf nb3 = load_bfrag(Btbase + 48 * K + kn, K, lane);

    acc[0] = wmma_bf16(a0, b0, acc[0]);
    acc[1] = wmma_bf16(a1, b0, acc[1]);
    acc[2] = wmma_bf16(a0, b1, acc[2]);
    acc[3] = wmma_bf16(a1, b1, acc[3]);
    acc[4] = wmma_bf16(a0, b2, acc[4]);
    acc[5] = wmma_bf16(a1, b2, acc[5]);
    acc[6] = wmma_bf16(a0, b3, acc[6]);
    acc[7] = wmma_bf16(a1, b3, acc[7]);

    a0 = na0; a1 = na1;
    b0 = nb0; b1 = nb1; b2 = nb2; b3 = nb3;
  }
}

// ---------------- QKV GEMM:  qkv = xb @ Wqkv, scattered into Q/K/Vt ----------------
// xb: [4096][1024] bf16 ; wt: [3072][1024] bf16 (Wqkv^T)
// Q,K: [B,H,S,HD] bf16 (Q pre-scaled) ; Vt: [B,H,HD,S] bf16
__global__ void qkv_gemm(const bf16* __restrict__ xb, const bf16* __restrict__ wt,
                         bf16* __restrict__ Qb, bf16* __restrict__ Kb,
                         bf16* __restrict__ Vtb) {
  int lane = threadIdx.x & 31;
  int wg = blockIdx.x * (blockDim.x >> 5) + (threadIdx.x >> 5);
  int tn = wg % 48;           // 3072/64 column tiles
  int tm = wg / 48;           // 4096/32 row tiles
  int r0 = tm * 32, c0 = tn * 64;

  v8f acc[8] = {};
  gemm_mainloop_32x64(xb + (size_t)r0 * DIM_, wt + (size_t)c0 * DIM_, lane, acc);

  int n = lane & 15, hl = lane >> 4;
#pragma unroll
  for (int ar = 0; ar < 2; ar++) {
#pragma unroll
    for (int t = 0; t < 4; t++) {
#pragma unroll
      for (int v = 0; v < 8; v++) {
        int rr = r0 + ar * 16 + 8 * hl + v;  // [0,4096) = b*S + s
        int cc = c0 + t * 16 + n;            // [0,3072) = which*1024 + h*64 + hd
        int b_    = rr >> 11;
        int s_    = rr & (S_ - 1);
        int which = cc >> 10;
        int h_    = (cc & 1023) >> 6;
        int hd    = cc & 63;
        float val = acc[ar * 4 + t][v];
        size_t qk_idx = ((size_t)(b_ * H_ + h_) * S_ + s_) * HD_ + hd;
        if (which == 0)       Qb[qk_idx] = (bf16)(val * SCALE_);
        else if (which == 1)  Kb[qk_idx] = (bf16)val;
        else Vtb[((size_t)(b_ * H_ + h_) * HD_ + hd) * S_ + s_] = (bf16)val;
      }
    }
  }
}

// ---------------- Flash attention ----------------
// 128 threads = 4 waves; each wave: 16 queries x full HD=64, streaming 32-key blocks.
__global__ void flash_attn(const bf16* __restrict__ Qb, const bf16* __restrict__ Kb,
                           const bf16* __restrict__ Vtb, bf16* __restrict__ Ob) {
  __shared__ __align__(16) float Sld[4][16][33];   // fp32 scores, padded
  __shared__ __align__(16) bf16  Pld[4][16][48];   // bf16 P, stride 48 (96B, 16B aligned)
  __shared__ float Arow[4][16];
  __shared__ float Lrow[4][16];

  int lane = threadIdx.x & 31;
  int w    = threadIdx.x >> 5;
  int bq   = blockIdx.x & 31;          // S/64 = 32 query blocks
  int bh   = blockIdx.x >> 5;          // [0, B*H)
  int b    = bh >> 4;
  int h    = bh & 15;

  const bf16* Qh  = Qb  + (size_t)(b * H_ + h) * S_ * HD_;
  const bf16* Kh  = Kb  + (size_t)(b * H_ + h) * S_ * HD_;
  const bf16* Vth = Vtb + (size_t)(b * H_ + h) * HD_ * S_;
  int q0 = bq * 64 + w * 16;

  // Q fragments (K-dim = HD split into two 32-chunks); SCALE already folded in.
  v16bf qa0 = load_afrag(Qh + (size_t)q0 * HD_ +  0, HD_, lane);
  v16bf qa1 = load_afrag(Qh + (size_t)q0 * HD_ + 32, HD_, lane);

  v8f o0 = {}, o1 = {}, o2 = {}, o3 = {};
  float m_prev = -1e30f, l_prev = 0.f;
  int r  = lane & 15;   // softmax row handled by this lane
  int sg = lane >> 4;   // which 16-col segment
  int n  = lane & 15;   // D-frag column
  int hl = lane >> 4;   // D-frag row half

  // pipelined K fragments: kc* holds the current key block, loaded one iter ahead
  v16bf kc00 = load_bfrag(Kh +  0 * HD_ +  0, HD_, lane);
  v16bf kc01 = load_bfrag(Kh +  0 * HD_ + 32, HD_, lane);
  v16bf kc10 = load_bfrag(Kh + 16 * HD_ +  0, HD_, lane);
  v16bf kc11 = load_bfrag(Kh + 16 * HD_ + 32, HD_, lane);

  for (int kb0 = 0; kb0 < S_; kb0 += 32) {
    int kbn = (kb0 + 32) & (S_ - 1);   // wrapped next block (unused values on last iter)

    // issue V loads (consumed after softmax) and next-K loads (consumed next iter)
    v16bf vf0 = load_bfrag(Vth + (size_t)( 0) * S_ + kb0, S_, lane);
    v16bf vf1 = load_bfrag(Vth + (size_t)(16) * S_ + kb0, S_, lane);
    v16bf vf2 = load_bfrag(Vth + (size_t)(32) * S_ + kb0, S_, lane);
    v16bf vf3 = load_bfrag(Vth + (size_t)(48) * S_ + kb0, S_, lane);
    v16bf kn00 = load_bfrag(Kh + (size_t)(kbn +  0) * HD_ +  0, HD_, lane);
    v16bf kn01 = load_bfrag(Kh + (size_t)(kbn +  0) * HD_ + 32, HD_, lane);
    v16bf kn10 = load_bfrag(Kh + (size_t)(kbn + 16) * HD_ +  0, HD_, lane);
    v16bf kn11 = load_bfrag(Kh + (size_t)(kbn + 16) * HD_ + 32, HD_, lane);

    // ---- scores: S[16q x 32k] = Q * K^T (scale folded into Q) ----
    v8f s0 = {}, s1 = {};
    s0 = wmma_bf16(qa0, kc00, s0);
    s0 = wmma_bf16(qa1, kc01, s0);
    s1 = wmma_bf16(qa0, kc10, s1);
    s1 = wmma_bf16(qa1, kc11, s1);

#pragma unroll
    for (int v = 0; v < 8; v++) {
      Sld[w][8 * hl + v][n]      = s0[v];
      Sld[w][8 * hl + v][16 + n] = s1[v];
    }
    __syncthreads();

    // ---- online softmax (lane pair L / L^16 covers one row) ----
    float sv[16];
    float mx = -1e30f;
#pragma unroll
    for (int j = 0; j < 16; j++) {
      sv[j] = Sld[w][r][sg * 16 + j];
      mx = fmaxf(mx, sv[j]);
    }
    mx = fmaxf(mx, __shfl_xor(mx, 16, 32));
    float m_new = fmaxf(m_prev, mx);
    float alpha = __expf(m_prev - m_new);
    float sum = 0.f;
#pragma unroll
    for (int j = 0; j < 16; j++) {
      float p = __expf(sv[j] - m_new);
      sum += p;
      Pld[w][r][sg * 16 + j] = (bf16)p;
    }
    sum += __shfl_xor(sum, 16, 32);
    l_prev = l_prev * alpha + sum;
    m_prev = m_new;
    if (sg == 0) Arow[w][r] = alpha;
    __syncthreads();

    // ---- rescale O accumulators, then O += P @ V ----
#pragma unroll
    for (int v = 0; v < 8; v++) {
      float al = Arow[w][8 * hl + v];
      o0[v] *= al; o1[v] *= al; o2[v] *= al; o3[v] *= al;
    }
    v16bf pa = load_afrag(&Pld[w][0][0], 48, lane);
    o0 = wmma_bf16(pa, vf0, o0);
    o1 = wmma_bf16(pa, vf1, o1);
    o2 = wmma_bf16(pa, vf2, o2);
    o3 = wmma_bf16(pa, vf3, o3);

    kc00 = kn00; kc01 = kn01; kc10 = kn10; kc11 = kn11;
  }

  if (sg == 0) Lrow[w][r] = l_prev;
  __syncthreads();

  // normalize and store bf16 attention output, row-major [B*S][DIM]
#pragma unroll
  for (int v = 0; v < 8; v++) {
    float inv = 1.0f / Lrow[w][8 * hl + v];
    int grow = b * S_ + q0 + 8 * hl + v;
    bf16* op = Ob + (size_t)grow * DIM_ + h * HD_;
    op[ 0 + n] = (bf16)(o0[v] * inv);
    op[16 + n] = (bf16)(o1[v] * inv);
    op[32 + n] = (bf16)(o2[v] * inv);
    op[48 + n] = (bf16)(o3[v] * inv);
  }
}

// ---------------- projection GEMM: out = Ob @ Wproj (fp32 out) ----------------
__global__ void proj_gemm(const bf16* __restrict__ Ob, const bf16* __restrict__ wpt,
                          float* __restrict__ out) {
  int lane = threadIdx.x & 31;
  int wg = blockIdx.x * (blockDim.x >> 5) + (threadIdx.x >> 5);
  int tn = wg % 16;           // 1024/64 col tiles
  int tm = wg / 16;           // 4096/32 row tiles
  int r0 = tm * 32, c0 = tn * 64;

  v8f acc[8] = {};
  gemm_mainloop_32x64(Ob + (size_t)r0 * DIM_, wpt + (size_t)c0 * DIM_, lane, acc);

  int n = lane & 15, hl = lane >> 4;
#pragma unroll
  for (int ar = 0; ar < 2; ar++) {
#pragma unroll
    for (int t = 0; t < 4; t++) {
#pragma unroll
      for (int v = 0; v < 8; v++) {
        int rr = r0 + ar * 16 + 8 * hl + v;
        int cc = c0 + t * 16 + n;
        out[(size_t)rr * DIM_ + cc] = acc[ar * 4 + t][v];
      }
    }
  }
}

// ---------------- launch ----------------
extern "C" void kernel_launch(void* const* d_in, const int* in_sizes, int n_in,
                              void* d_out, int out_size, void* d_ws, size_t ws_size,
                              hipStream_t stream) {
  const float* x     = (const float*)d_in[0];   // [2,2048,1024]
  const float* Wqkv  = (const float*)d_in[1];   // [1024,3072]
  const float* Wproj = (const float*)d_in[2];   // [1024,1024]
  float* out = (float*)d_out;

  char* ws = (char*)d_ws;
  // workspace layout (48 MB total)
  bf16* xb     = (bf16*)(ws);                    //  8 MB  x in bf16 [4096][1024]
  bf16* wqkvt  = (bf16*)(ws +  8388608);         //  6 MB  Wqkv^T [3072][1024]
  bf16* wprojt = (bf16*)(ws + 14680064);         //  2 MB  Wproj^T [1024][1024]
  bf16* Qb     = (bf16*)(ws + 16777216);         //  8 MB  [B,H,S,HD] (pre-scaled)
  bf16* Kb     = (bf16*)(ws + 25165824);         //  8 MB  [B,H,S,HD]
  bf16* Vtb    = (bf16*)(ws + 33554432);         //  8 MB  [B,H,HD,S]
  bf16* Ob     = (bf16*)(ws + 41943040);         //  8 MB  attn out [4096][1024]

  const int n_x = B_ * S_ * DIM_;                // 4,194,304
  cvt_f32_bf16<<<(n_x + 255) / 256, 256, 0, stream>>>(x, xb, n_x);
  cvt_transpose_bf16<<<(3 * DIM_ * DIM_ + 255) / 256, 256, 0, stream>>>(Wqkv, wqkvt, DIM_, 3 * DIM_);
  cvt_transpose_bf16<<<(DIM_ * DIM_ + 255) / 256, 256, 0, stream>>>(Wproj, wprojt, DIM_, DIM_);

  // 4096/32 * 3072/64 = 6144 wave-tiles, 8 waves per 256-thread block
  qkv_gemm<<<768, 256, 0, stream>>>(xb, wqkvt, Qb, Kb, Vtb);

  // B*H*(S/64) = 1024 blocks, 4 waves each (one 16-query tile per wave)
  flash_attn<<<1024, 128, 0, stream>>>(Qb, Kb, Vtb, Ob);

  // 4096/32 * 1024/64 = 2048 wave-tiles
  proj_gemm<<<256, 256, 0, stream>>>(Ob, wprojt, out);
}
